// GraphAttentionNetwork_38482906972561
// MI455X (gfx1250) — matrix operand order repaired and checked by reading, not compiled, not run on hardware
//
#include <hip/hip_runtime.h>
#include <hip/hip_fp16.h>
#include <stdint.h>

typedef __attribute__((ext_vector_type(16))) _Float16 v16h;
typedef __attribute__((ext_vector_type(8)))  _Float16 v8h;
typedef __attribute__((ext_vector_type(8)))  float    v8f;

#define N_NODES 50000
#define BATCH   4
#define F_IN    128
#define HID     64
#define HEADS   8
#define OUTD    64
#define ROWS    (BATCH * N_NODES)   // 200000
#define C1      (HEADS * HID)       // 512
#define SLOPE   0.2f

// ---------------- helpers ----------------
__device__ __forceinline__ float atomicMaxF(float* addr, float val) {
  // total-order float max via int/uint monotonic mapping (no NaN inputs)
  return (val >= 0.f)
    ? __int_as_float(atomicMax((int*)addr, __float_as_int(val)))
    : __uint_as_float(atomicMin((unsigned int*)addr, __float_as_uint(val)));
}

__device__ __forceinline__ float lrelu(float x) { return x > 0.f ? x : SLOPE * x; }

// ---------------- small utility kernels ----------------
__global__ void k_f32_to_f16(const float* __restrict__ src, _Float16* __restrict__ dst, long long n) {
  long long i = (long long)blockIdx.x * blockDim.x + threadIdx.x;
  if (i < n) dst[i] = (_Float16)src[i];
}

__global__ void k_fill_f32(float* __restrict__ p, float v, long long n) {
  long long i = (long long)blockIdx.x * blockDim.x + threadIdx.x;
  if (i < n) p[i] = v;
}

__global__ void k_fill_bias(float* __restrict__ p, const float* __restrict__ bias,
                            long long n, int blen) {
  long long i = (long long)blockIdx.x * blockDim.x + threadIdx.x;
  if (i < n) p[i] = bias[i % blen];
}

// Pack W1[h][k][d] (h<8,k<128,d<64) into B-fragment order for 32x16 K-tiles.
// packed index p: i = p&15 (half within lane), lane = (p>>4)&31, tile t = p>>9,
// kt = t % (K/32), nt = t / (K/32); k = kt*32 + (lane>>4)*16 + i; n = nt*16 + (lane&15)
__global__ void k_pack_w1(const float* __restrict__ W1, _Float16* __restrict__ wp) {
  const int K = F_IN;
  int p = blockIdx.x * blockDim.x + threadIdx.x;
  if (p >= K * C1) return;
  int i = p & 15, lane = (p >> 4) & 31, t = p >> 9;
  int kt = t % (K / 32), nt = t / (K / 32);
  int k = kt * 32 + (lane >> 4) * 16 + i;
  int n = nt * 16 + (lane & 15);
  int h = n >> 6, d = n & 63;
  wp[p] = (_Float16)W1[((size_t)h * K + k) * HID + d];
}

__global__ void k_pack_w2(const float* __restrict__ W2, _Float16* __restrict__ wp) {
  const int K = C1;
  int p = blockIdx.x * blockDim.x + threadIdx.x;
  if (p >= K * OUTD) return;
  int i = p & 15, lane = (p >> 4) & 31, t = p >> 9;
  int kt = t % (K / 32), nt = t / (K / 32);
  int k = kt * 32 + (lane >> 4) * 16 + i;
  int n = nt * 16 + (lane & 15);
  wp[p] = (_Float16)W2[(size_t)k * OUTD + n];
}

// ---------------- WMMA GEMM with LDS-staged, register-resident B ----------------
// C[M,NC] = A[M,K](f16 row-major) * Bp(pre-packed fragments), fp32 accum.
// Each block owns one column group (NT 16-wide tiles) and stages all
// KSTEPS*NT B fragments into LDS once; its 8 waves grid-stride over M tiles.
// The loop-invariant LDS reads get hoisted by the compiler into registers
// (128 VGPRs of B); __launch_bounds__(256,1) grants the full register file
// so nothing spills. Inner loop: 2 global b128 (A) -> NT v_wmma.
// SH/SF select f16/f32 output at compile time (branch-free epilogue).
template <int KSTEPS, int NT, bool SH, bool SF>
__global__ __launch_bounds__(256, 1)
void k_gemm_lds(const _Float16* __restrict__ A, const _Float16* __restrict__ Bp,
                _Float16* __restrict__ outH, float* __restrict__ outF,
                int M, int NC, int nGroups) {
  constexpr int NFRAG = KSTEPS * NT;
  __shared__ __align__(16) unsigned char smemB[NFRAG * 1536];

  const int lane = threadIdx.x & 31;
  const int wave = threadIdx.x >> 5;
  const int group = blockIdx.x % nGroups;          // column group (NT*16 cols)
  const int blockSlot = blockIdx.x / nGroups;
  const int blocksPerGroup = gridDim.x / nGroups;
  const int tilesM = M >> 4;
  const int K = KSTEPS * 32;

  // stage this group's B fragments into LDS (once per block)
  for (int f = wave; f < NFRAG; f += 8) {
    const _Float16* src = Bp + ((size_t)group * NFRAG + f) * 512 + lane * 16;
    v8h lo = *(const v8h*)src;
    v8h hi = *(const v8h*)(src + 8);
    *(v8h*)(smemB + f * 1536 + lane * 48) = lo;
    *(v8h*)(smemB + f * 1536 + lane * 48 + 16) = hi;
  }
  __syncthreads();

  const int rowAdd = (lane >> 4) << 3;   // C row offset within tile
  const int colIn  = lane & 15;
  const int mtStride = blocksPerGroup * 8;

  for (int mt = blockSlot * 8 + wave; mt < tilesM; mt += mtStride) {
    const int row0 = mt << 4;
    // A fragment: lane L -> row row0 + L%16, halves {hb..hb+15}, hb=(L/16)*16
    const _Float16* aptr = A + (size_t)(row0 + colIn) * K + ((lane >> 4) << 4);

    v8f acc[NT];
#pragma unroll
    for (int j = 0; j < NT; ++j) acc[j] = (v8f){};

#pragma unroll
    for (int kt = 0; kt < KSTEPS; ++kt) {
      v8h lo = *(const v8h*)(aptr + kt * 32);
      v8h hi = *(const v8h*)(aptr + kt * 32 + 8);
      v16h af = __builtin_shufflevector(lo, hi, 0,1,2,3,4,5,6,7,8,9,10,11,12,13,14,15);
#pragma unroll
      for (int j = 0; j < NT; ++j) {
        const unsigned char* bp = smemB + (j * KSTEPS + kt) * 1536 + lane * 48;
        v8h bl = *(const v8h*)bp;
        v8h bh = *(const v8h*)(bp + 16);
        v16h bf = __builtin_shufflevector(bl, bh, 0,1,2,3,4,5,6,7,8,9,10,11,12,13,14,15);
        acc[j] = __builtin_amdgcn_wmma_f32_16x16x32_f16(false, af, false, bf,
                                                        (short)0, acc[j], false, false);
      }
    }

    // C/D layout: lane L -> col L%16 ; VGPR r -> row r + 8*(L/16)
#pragma unroll
    for (int r = 0; r < 8; ++r) {
      size_t base = (size_t)(row0 + rowAdd + r) * NC + group * (NT * 16) + colIn;
#pragma unroll
      for (int j = 0; j < NT; ++j) {
        if constexpr (SH) outH[base + j * 16] = (_Float16)acc[j][r];
        if constexpr (SF) outF[base + j * 16] = acc[j][r];
      }
    }
  }
}

// ---------------- attention scores ----------------
template <typename T>
__global__ void k_scores(const T* __restrict__ feat, const float* __restrict__ a_s,
                         const float* __restrict__ a_d, float* __restrict__ es,
                         float* __restrict__ ed, int rows, int H) {
  long long gid = (long long)blockIdx.x * blockDim.x + threadIdx.x;
  if (gid >= (long long)rows * H) return;
  int h = (int)(gid % H);
  long long row = gid / H;
  const T* f = feat + ((size_t)row * H + h) * HID;
  const float* as = a_s + h * HID;
  const float* ad = a_d + h * HID;
  float s = 0.f, d = 0.f;
#pragma unroll 8
  for (int i = 0; i < HID; ++i) {
    float v = (float)f[i];
    s = fmaf(v, as[i], s);
    d = fmaf(v, ad[i], d);
  }
  es[gid] = s;
  ed[gid] = d;
}

// ---------------- edge passes ----------------
__device__ __forceinline__ void edge_decode(long long t, const int* ei, int E,
                                            int& e, int& b, int& s, int& d) {
  b = (int)(t % BATCH);
  e = (int)(t / BATCH);
  if (e < E) { s = ei[e]; d = ei[E + e]; } else { s = d = e - E; }
}

__global__ void k_edge_max(const int* __restrict__ ei, int E, int Etot,
                           const float* __restrict__ es, const float* __restrict__ ed,
                           float* __restrict__ m, int H) {
  long long gid = (long long)blockIdx.x * blockDim.x + threadIdx.x;
  if (gid >= (long long)Etot * BATCH * H) return;
  int h = (int)(gid % H);
  int e, b, s, d;
  edge_decode(gid / H, ei, E, e, b, s, d);
  float sc = lrelu(es[((size_t)b * N_NODES + s) * H + h] + ed[((size_t)b * N_NODES + d) * H + h]);
  atomicMaxF(&m[((size_t)b * N_NODES + d) * H + h], sc);
}

// ex = exp(score - max) stored per (e,b,h); den accumulated atomically
__global__ void k_edge_ex(const int* __restrict__ ei, int E, int Etot,
                          const float* __restrict__ es, const float* __restrict__ ed,
                          const float* __restrict__ m, float* __restrict__ den,
                          float* __restrict__ alpha, int H) {
  long long gid = (long long)blockIdx.x * blockDim.x + threadIdx.x;
  if (gid >= (long long)Etot * BATCH * H) return;
  int h = (int)(gid % H);
  int e, b, s, d;
  edge_decode(gid / H, ei, E, e, b, s, d);
  size_t di = ((size_t)b * N_NODES + d) * H + h;
  float sc = lrelu(es[((size_t)b * N_NODES + s) * H + h] + ed[di]);
  float ex = expf(sc - m[di]);
  alpha[gid] = ex;
  atomicAdd(&den[di], ex);
}

__global__ void k_edge_norm(const int* __restrict__ ei, int E, int Etot,
                            const float* __restrict__ den, float* __restrict__ alpha, int H) {
  long long gid = (long long)blockIdx.x * blockDim.x + threadIdx.x;
  if (gid >= (long long)Etot * BATCH * H) return;
  int h = (int)(gid % H);
  int e, b, s, d;
  edge_decode(gid / H, ei, E, e, b, s, d);
  alpha[gid] /= den[((size_t)b * N_NODES + d) * H + h];
}

// feature-parallel aggregation: consecutive lanes take consecutive features,
// so feature loads and f32 atomic adds are wave-contiguous (coalesced).
template <typename T, int CF>
__global__ void k_edge_agg_flat(const int* __restrict__ ei, int E, int Etot,
                                const float* __restrict__ alpha,
                                const T* __restrict__ feat, float* __restrict__ out) {
  long long gid = (long long)blockIdx.x * blockDim.x + threadIdx.x;
  if (gid >= (long long)Etot * BATCH * CF) return;
  int c = (int)(gid % CF);
  int e, b, s, d;
  edge_decode(gid / CF, ei, E, e, b, s, d);
  float a = alpha[((size_t)e * BATCH + b) * (CF / 64) + (c >> 6)];
  atomicAdd(&out[((size_t)b * N_NODES + d) * CF + c],
            a * (float)feat[((size_t)b * N_NODES + s) * CF + c]);
}

// ---------------- epilogue: ELU + log_softmax, one wave per row ----------------
__global__ void k_elu_logsoftmax(float* __restrict__ o, int rows) {
  int lane = threadIdx.x & 31;
  long long row = (long long)blockIdx.x * (blockDim.x >> 5) + (threadIdx.x >> 5);
  if (row >= rows) return;
  float* p = o + (size_t)row * OUTD;
  float x0 = p[lane], x1 = p[lane + 32];
  x0 = x0 > 0.f ? x0 : expm1f(x0);
  x1 = x1 > 0.f ? x1 : expm1f(x1);
  float mx = fmaxf(x0, x1);
#pragma unroll
  for (int off = 16; off > 0; off >>= 1) mx = fmaxf(mx, __shfl_xor(mx, off, 32));
  float s = expf(x0 - mx) + expf(x1 - mx);
#pragma unroll
  for (int off = 16; off > 0; off >>= 1) s += __shfl_xor(s, off, 32);
  float ls = logf(s) + mx;
  p[lane] = x0 - ls;
  p[lane + 32] = x1 - ls;
}

// ---------------- launch ----------------
static inline unsigned gridFor(long long n, int bs) { return (unsigned)((n + bs - 1) / bs); }

extern "C" void kernel_launch(void* const* d_in, const int* in_sizes, int n_in,
                              void* d_out, int out_size, void* d_ws, size_t ws_size,
                              hipStream_t stream) {
  (void)n_in; (void)out_size; (void)ws_size;
  const float* x   = (const float*)d_in[0];
  const int*   ei  = (const int*)d_in[1];
  const float* W1  = (const float*)d_in[2];
  const float* as1 = (const float*)d_in[3];
  const float* ad1 = (const float*)d_in[4];
  const float* b1  = (const float*)d_in[5];
  const float* W2  = (const float*)d_in[6];
  const float* as2 = (const float*)d_in[7];
  const float* ad2 = (const float*)d_in[8];
  const float* b2  = (const float*)d_in[9];
  float* out = (float*)d_out;
  const int E    = in_sizes[1] / 2;
  const int Etot = E + N_NODES;

  char* ws = (char*)d_ws;
  size_t off = 0;
  auto alloc = [&](size_t bytes) { size_t o = off; off += (bytes + 255) & ~(size_t)255; return o; };

  _Float16* xf16   = (_Float16*)(ws + alloc((size_t)ROWS * F_IN * 2)); // reused as h2 f32 (same size)
  _Float16* h1f16  = (_Float16*)(ws + alloc((size_t)ROWS * C1 * 2));   // reused as out1 f16
  float*    out1   = (float*)(ws + alloc((size_t)ROWS * C1 * 4));
  float*    es1    = (float*)(ws + alloc((size_t)ROWS * HEADS * 4));
  float*    ed1    = (float*)(ws + alloc((size_t)ROWS * HEADS * 4));
  float*    m1     = (float*)(ws + alloc((size_t)ROWS * HEADS * 4));
  float*    den1   = (float*)(ws + alloc((size_t)ROWS * HEADS * 4));
  float*    es2    = (float*)(ws + alloc((size_t)ROWS * 4));
  float*    ed2    = (float*)(ws + alloc((size_t)ROWS * 4));
  float*    m2     = (float*)(ws + alloc((size_t)ROWS * 4));
  float*    den2   = (float*)(ws + alloc((size_t)ROWS * 4));
  float*    alpha1 = (float*)(ws + alloc((size_t)Etot * BATCH * HEADS * 4));
  float*    alpha2 = (float*)(ws + alloc((size_t)Etot * BATCH * 4));
  _Float16* wp1    = (_Float16*)(ws + alloc((size_t)F_IN * C1 * 2));
  _Float16* wp2    = (_Float16*)(ws + alloc((size_t)C1 * OUTD * 2));
  float*    h2     = (float*)xf16;  // layer-2 projection output reuses x_f16 region

  const int BS = 256;
  const int GEMM_BLOCKS = 1024;     // divisible by 8 and 4 column groups

  // --- prep: casts, weight packing, initializations ---
  k_f32_to_f16<<<gridFor((long long)ROWS * F_IN, BS), BS, 0, stream>>>(x, xf16, (long long)ROWS * F_IN);
  k_pack_w1<<<gridFor(F_IN * C1, BS), BS, 0, stream>>>(W1, wp1);
  k_pack_w2<<<gridFor(C1 * OUTD, BS), BS, 0, stream>>>(W2, wp2);
  k_fill_f32<<<gridFor((long long)ROWS * HEADS, BS), BS, 0, stream>>>(m1, -3.4e38f, (long long)ROWS * HEADS);
  k_fill_f32<<<gridFor((long long)ROWS * HEADS, BS), BS, 0, stream>>>(den1, 0.f, (long long)ROWS * HEADS);
  k_fill_f32<<<gridFor(ROWS, BS), BS, 0, stream>>>(m2, -3.4e38f, ROWS);
  k_fill_f32<<<gridFor(ROWS, BS), BS, 0, stream>>>(den2, 0.f, ROWS);
  k_fill_bias<<<gridFor((long long)ROWS * C1, BS), BS, 0, stream>>>(out1, b1, (long long)ROWS * C1, C1);
  k_fill_bias<<<gridFor((long long)ROWS * OUTD, BS), BS, 0, stream>>>(out, b2, (long long)ROWS * OUTD, OUTD);

  // --- layer 1: projection (WMMA, B staged in LDS), scores, softmax-aggregate ---
  k_gemm_lds<4, 4, true, false><<<GEMM_BLOCKS, BS, 0, stream>>>(xf16, wp1, h1f16, nullptr, ROWS, C1, C1 / 64);
  k_scores<_Float16><<<gridFor((long long)ROWS * HEADS, BS), BS, 0, stream>>>(h1f16, as1, ad1, es1, ed1, ROWS, HEADS);
  {
    long long tot = (long long)Etot * BATCH * HEADS;
    k_edge_max<<<gridFor(tot, BS), BS, 0, stream>>>(ei, E, Etot, es1, ed1, m1, HEADS);
    k_edge_ex<<<gridFor(tot, BS), BS, 0, stream>>>(ei, E, Etot, es1, ed1, m1, den1, alpha1, HEADS);
    k_edge_norm<<<gridFor(tot, BS), BS, 0, stream>>>(ei, E, Etot, den1, alpha1, HEADS);
    k_edge_agg_flat<_Float16, C1><<<gridFor((long long)Etot * BATCH * C1, BS), BS, 0, stream>>>(
        ei, E, Etot, alpha1, h1f16, out1);
  }

  // --- layer 2: cast, projection (WMMA), scores, softmax-aggregate into d_out ---
  k_f32_to_f16<<<gridFor((long long)ROWS * C1, BS), BS, 0, stream>>>(out1, h1f16, (long long)ROWS * C1);
  k_gemm_lds<16, 1, false, true><<<GEMM_BLOCKS, BS, 0, stream>>>(h1f16, wp2, nullptr, h2, ROWS, OUTD, OUTD / 16);
  k_scores<float><<<gridFor(ROWS, BS), BS, 0, stream>>>(h2, as2, ad2, es2, ed2, ROWS, 1);
  {
    long long tot = (long long)Etot * BATCH;
    k_edge_max<<<gridFor(tot, BS), BS, 0, stream>>>(ei, E, Etot, es2, ed2, m2, 1);
    k_edge_ex<<<gridFor(tot, BS), BS, 0, stream>>>(ei, E, Etot, es2, ed2, m2, den2, alpha2, 1);
    k_edge_norm<<<gridFor(tot, BS), BS, 0, stream>>>(ei, E, Etot, den2, alpha2, 1);
    k_edge_agg_flat<float, OUTD><<<gridFor((long long)Etot * BATCH * OUTD, BS), BS, 0, stream>>>(
        ei, E, Etot, alpha2, h2, out);
  }

  // --- epilogue ---
  k_elu_logsoftmax<<<gridFor((long long)ROWS * 32, BS), BS, 0, stream>>>(out, ROWS);
}
